// Attention_7516192768567
// MI455X (gfx1250) — matrix operand order repaired and checked by reading, not compiled, not run on hardware
//
#include <hip/hip_runtime.h>
#include <hip/hip_bf16.h>
#include <math.h>

// ---------- types ----------
typedef __bf16 bf16_t;
typedef __attribute__((ext_vector_type(16))) __bf16 v16bf;
typedef __attribute__((ext_vector_type(8)))  __bf16 v8bf;
typedef __attribute__((ext_vector_type(4)))  __bf16 v4bf;
typedef __attribute__((ext_vector_type(8)))  float  v8f;
typedef __attribute__((ext_vector_type(4)))  float  v4f;

#define HEADS    8
#define DIM_HEAD 64
#define SEQ      512
#define DIM      512
#define BP       32          // b*p = 2*16
#define ATT_SCALE 0.125f     // 64^-0.5

#define WMMA_BF16(a, b, c) \
  __builtin_amdgcn_wmma_f32_16x16x32_bf16(false, (a), false, (b), (short)0, (c), false, false)

static __device__ __forceinline__ v16bf pack16(v8bf lo, v8bf hi) {
  return __builtin_shufflevector(lo, hi, 0,1,2,3,4,5,6,7,8,9,10,11,12,13,14,15);
}

// A-fragment (16x32, bf16) from a row-major bf16 matrix. rowPtr = &A[m][0] for
// this lane's row m = tileRow + (lane&15). hf = lane>>4.
// Element e -> K = kBase + hf*16 + e   (same k order as load_b)
static __device__ __forceinline__ v16bf load_a_bf16(const bf16_t* rowPtr, int kBase, int hf) {
  v8bf lo = *(const v8bf*)(rowPtr + kBase + hf * 16);
  v8bf hi = *(const v8bf*)(rowPtr + kBase + hf * 16 + 8);
  return pack16(lo, hi);
}

// B-fragment (32x16, bf16) from a [N][K] bf16 matrix (one memory row per output
// column). colRowPtr = &Bt[n][0] for this lane's column n = tileCol + (lane&15).
// Element e -> K = kBase + hf*16 + e   (contiguous k order, same as load_a_bf16)
static __device__ __forceinline__ v16bf load_b(const bf16_t* colRowPtr, int kBase, int hf) {
  v8bf lo = *(const v8bf*)(colRowPtr + kBase + hf * 16);
  v8bf hi = *(const v8bf*)(colRowPtr + kBase + hf * 16 + 8);
  return pack16(lo, hi);
}

// ---------- kernel 0a: f32 [K][N] -> bf16 [N][K] ----------
__global__ void k_convert_transpose(const float* __restrict__ W, bf16_t* __restrict__ Wt,
                                    int K, int N) {
  int idx = blockIdx.x * blockDim.x + threadIdx.x;
  if (idx >= K * N) return;
  int k = idx / N;
  int n = idx - k * N;
  Wt[(size_t)n * K + k] = (bf16_t)W[idx];
}

// ---------- kernel 0b: f32 -> bf16 elementwise (x), 4-wide ----------
__global__ void k_convert_x(const float* __restrict__ x, bf16_t* __restrict__ xb, int n) {
  int i4 = (blockIdx.x * blockDim.x + threadIdx.x) * 4;
  if (i4 >= n) return;
  v4f f = *(const v4f*)(x + i4);
  v4bf o;
#pragma unroll
  for (int e = 0; e < 4; ++e) o[e] = (bf16_t)f[e];
  *(v4bf*)(xb + i4) = o;
}

// ---------- kernel 1: QKV projection ----------
// grid = (4 Mtiles, 6 Ntiles, 32 bp), block = 256 (8 waves, 64x64 tile each)
// Block tile = 128(M) x 256(N); wave layout 2(M) x 4(N).
__global__ __launch_bounds__(256) void k_qkv_gemm(const bf16_t* __restrict__ xb,
                                                  const bf16_t* __restrict__ Wt, // [1536][512]
                                                  bf16_t* __restrict__ q,
                                                  bf16_t* __restrict__ kmat,
                                                  bf16_t* __restrict__ vT) {
  const int bp   = blockIdx.z;
  const int lane = threadIdx.x & 31;
  const int wave = threadIdx.x >> 5;
  const int cl = lane & 15, hf = lane >> 4;
  const int M0 = blockIdx.x * 128 + (wave >> 2) * 64;
  const int N0 = blockIdx.y * 256 + (wave & 3) * 64;
  const bf16_t* A = xb + (size_t)bp * SEQ * DIM;

  v8f zero = {};
  v8f acc[4][4];
#pragma unroll
  for (int mt = 0; mt < 4; ++mt)
#pragma unroll
    for (int nt = 0; nt < 4; ++nt) acc[mt][nt] = zero;

  for (int kb = 0; kb < DIM; kb += 32) {
    if (kb + 32 < DIM)  // hint next A strip toward L2/WGP$
      __builtin_prefetch(A + (size_t)(M0 + cl) * DIM + kb + 32, 0, 1);
    v16bf a[4], b[4];
#pragma unroll
    for (int mt = 0; mt < 4; ++mt)
      a[mt] = load_a_bf16(A + (size_t)(M0 + mt * 16 + cl) * DIM, kb, hf);
#pragma unroll
    for (int nt = 0; nt < 4; ++nt)
      b[nt] = load_b(Wt + (size_t)(N0 + nt * 16 + cl) * DIM, kb, hf);
#pragma unroll
    for (int mt = 0; mt < 4; ++mt)
#pragma unroll
      for (int nt = 0; nt < 4; ++nt)
        acc[mt][nt] = WMMA_BF16(a[mt], b[nt], acc[mt][nt]);
  }

  // D layout: VGPR r, lane -> row = r + 8*hf, col = cl. Split columns into q/k/vT.
#pragma unroll
  for (int nt = 0; nt < 4; ++nt) {
    const int cBase = N0 + nt * 16;     // lane-uniform
    const int which = cBase >> 9;       // 0=q 1=k 2=v
    const int h  = (cBase >> 6) & 7;
    const int d  = (cBase & 63) + cl;
#pragma unroll
    for (int mt = 0; mt < 4; ++mt) {
#pragma unroll
      for (int r = 0; r < 8; ++r) {
        const int row = M0 + mt * 16 + r + hf * 8;
        bf16_t v = (bf16_t)acc[mt][nt][r];
        if (which == 0)
          q[(((size_t)bp * HEADS + h) * SEQ + row) * DIM_HEAD + d] = v;
        else if (which == 1)
          kmat[(((size_t)bp * HEADS + h) * SEQ + row) * DIM_HEAD + d] = v;
        else
          vT[(((size_t)bp * HEADS + h) * DIM_HEAD + d) * SEQ + row] = v;
      }
    }
  }
}

// ---------- kernel 2: flash attention, one wave per 16-query tile ----------
// grid = (32 row tiles, 8 heads, 32 bp), block = 32 (one wave)
__global__ __launch_bounds__(32) void k_attn(const bf16_t* __restrict__ q,
                                             const bf16_t* __restrict__ kmat,
                                             const bf16_t* __restrict__ vT,
                                             bf16_t* __restrict__ ao) {
  const int M0 = blockIdx.x * 16;
  const int h  = blockIdx.y;
  const int bp = blockIdx.z;
  const int lane = threadIdx.x;
  const int cl = lane & 15, hf = lane >> 4;

  const bf16_t* qb = q    + (((size_t)bp * HEADS + h) * SEQ) * DIM_HEAD;
  const bf16_t* kb = kmat + (((size_t)bp * HEADS + h) * SEQ) * DIM_HEAD;
  const bf16_t* vb = vT   + (((size_t)bp * HEADS + h) * DIM_HEAD) * SEQ;

  // Q tile (16x64) held in registers as two A fragments (d 0..31, 32..63)
  v16bf qa[2];
#pragma unroll
  for (int s = 0; s < 2; ++s)
    qa[s] = load_a_bf16(qb + (size_t)(M0 + cl) * DIM_HEAD, s * 32, hf);

  v8f zero = {};
  v8f o[4];
#pragma unroll
  for (int nt = 0; nt < 4; ++nt) o[nt] = zero;
  float mrow[8], lrow[8];
#pragma unroll
  for (int r = 0; r < 8; ++r) { mrow[r] = -1e30f; lrow[r] = 0.0f; }

  __shared__ alignas(16) bf16_t pLds[16 * 32];   // D-layout -> A-layout staging

  for (int mb = 0; mb < SEQ; mb += 32) {         // 32 keys per block
    // S = Q K^T for key columns [mb, mb+32)
    v8f s0 = zero, s1 = zero;
#pragma unroll
    for (int s = 0; s < 2; ++s) {
      v16bf b0 = load_b(kb + (size_t)(mb + cl) * DIM_HEAD, s * 32, hf);
      v16bf b1 = load_b(kb + (size_t)(mb + 16 + cl) * DIM_HEAD, s * 32, hf);
      s0 = WMMA_BF16(qa[s], b0, s0);
      s1 = WMMA_BF16(qa[s], b1, s1);
    }
    // Online softmax. Row r (this lane half) spans 16 lanes -> XOR reduce.
#pragma unroll
    for (int r = 0; r < 8; ++r) {
      float t0 = s0[r] * ATT_SCALE, t1 = s1[r] * ATT_SCALE;
      float rm = fmaxf(t0, t1);
#pragma unroll
      for (int off = 1; off < 16; off <<= 1)
        rm = fmaxf(rm, __shfl_xor(rm, off, 32));
      float newm  = fmaxf(mrow[r], rm);
      float alpha = __expf(mrow[r] - newm);
      mrow[r] = newm;
      float p0 = __expf(t0 - newm);
      float p1 = __expf(t1 - newm);
      float rs = p0 + p1;
#pragma unroll
      for (int off = 1; off < 16; off <<= 1)
        rs += __shfl_xor(rs, off, 32);
      lrow[r] = lrow[r] * alpha + rs;
#pragma unroll
      for (int nt = 0; nt < 4; ++nt) o[nt][r] = o[nt][r] * alpha;
      const int prow = r + hf * 8;
      pLds[prow * 32 + cl]      = (bf16_t)p0;
      pLds[prow * 32 + 16 + cl] = (bf16_t)p1;
    }
    __syncthreads();
    // Reload P as an A fragment (16x32) and accumulate O += P * V
    v16bf pa = load_a_bf16(pLds + cl * 32, 0, hf);
#pragma unroll
    for (int nt = 0; nt < 4; ++nt) {
      v16bf bv = load_b(vb + (size_t)(nt * 16 + cl) * SEQ, mb, hf);
      o[nt] = WMMA_BF16(pa, bv, o[nt]);
    }
    __syncthreads();
  }

  // Normalize and store to attn_out[bp][n][h*64 + d] (bf16)
#pragma unroll
  for (int r = 0; r < 8; ++r) {
    float inv = 1.0f / lrow[r];
    const int row = M0 + r + hf * 8;
#pragma unroll
    for (int nt = 0; nt < 4; ++nt)
      ao[((size_t)bp * SEQ + row) * DIM + h * DIM_HEAD + nt * 16 + cl] =
          (bf16_t)(o[nt][r] * inv);
  }
}

// ---------- kernel 3: output projection + bias ----------
// grid = (4, 2, 32), block = 256 (8 waves, 64x64 tile each)
__global__ __launch_bounds__(256) void k_out_gemm(const bf16_t* __restrict__ ao,
                                                  const bf16_t* __restrict__ Wt, // [512][512]
                                                  const float* __restrict__ bias,
                                                  float* __restrict__ out) {
  const int bp   = blockIdx.z;
  const int lane = threadIdx.x & 31;
  const int wave = threadIdx.x >> 5;
  const int cl = lane & 15, hf = lane >> 4;
  const int M0 = blockIdx.x * 128 + (wave >> 2) * 64;
  const int N0 = blockIdx.y * 256 + (wave & 3) * 64;
  const bf16_t* A = ao + (size_t)bp * SEQ * DIM;

  v8f zero = {};
  v8f acc[4][4];
#pragma unroll
  for (int mt = 0; mt < 4; ++mt)
#pragma unroll
    for (int nt = 0; nt < 4; ++nt) acc[mt][nt] = zero;

  for (int kb = 0; kb < DIM; kb += 32) {
    v16bf a[4], b[4];
#pragma unroll
    for (int mt = 0; mt < 4; ++mt)
      a[mt] = load_a_bf16(A + (size_t)(M0 + mt * 16 + cl) * DIM, kb, hf);
#pragma unroll
    for (int nt = 0; nt < 4; ++nt)
      b[nt] = load_b(Wt + (size_t)(N0 + nt * 16 + cl) * DIM, kb, hf);
#pragma unroll
    for (int mt = 0; mt < 4; ++mt)
#pragma unroll
      for (int nt = 0; nt < 4; ++nt)
        acc[mt][nt] = WMMA_BF16(a[mt], b[nt], acc[mt][nt]);
  }

#pragma unroll
  for (int nt = 0; nt < 4; ++nt) {
    const int c = N0 + nt * 16 + cl;
    const float bv = bias[c];
#pragma unroll
    for (int mt = 0; mt < 4; ++mt)
#pragma unroll
      for (int r = 0; r < 8; ++r) {
        const int row = M0 + mt * 16 + r + hf * 8;
        out[((size_t)bp * SEQ + row) * DIM + c] = acc[mt][nt][r] + bv;
      }
  }
}

// ---------- host launch ----------
extern "C" void kernel_launch(void* const* d_in, const int* in_sizes, int n_in,
                              void* d_out, int out_size, void* d_ws, size_t ws_size,
                              hipStream_t stream) {
  (void)in_sizes; (void)n_in; (void)out_size; (void)ws_size;
  const float* x     = (const float*)d_in[0];
  const float* W_qkv = (const float*)d_in[1];
  const float* W_out = (const float*)d_in[2];
  const float* b_out = (const float*)d_in[3];
  float* out = (float*)d_out;

  char* ws = (char*)d_ws;
  size_t off = 0;
  auto alloc = [&](size_t bytes) -> char* {
    char* p = ws + off;
    off += (bytes + 255) & ~(size_t)255;
    return p;
  };
  bf16_t* WqkvT = (bf16_t*)alloc((size_t)1536 * 512 * sizeof(bf16_t));
  bf16_t* WoutT = (bf16_t*)alloc((size_t)512 * 512 * sizeof(bf16_t));
  bf16_t* xb    = (bf16_t*)alloc((size_t)BP * SEQ * DIM * sizeof(bf16_t));
  bf16_t* q     = (bf16_t*)alloc((size_t)BP * HEADS * SEQ * DIM_HEAD * sizeof(bf16_t));
  bf16_t* kmat  = (bf16_t*)alloc((size_t)BP * HEADS * SEQ * DIM_HEAD * sizeof(bf16_t));
  bf16_t* vT    = (bf16_t*)alloc((size_t)BP * HEADS * DIM_HEAD * SEQ * sizeof(bf16_t));
  bf16_t* ao    = (bf16_t*)alloc((size_t)BP * SEQ * DIM * sizeof(bf16_t));

  const int nx = BP * SEQ * DIM;
  k_convert_transpose<<<(512 * 1536 + 255) / 256, 256, 0, stream>>>(W_qkv, WqkvT, 512, 1536);
  k_convert_transpose<<<(512 * 512 + 255) / 256, 256, 0, stream>>>(W_out, WoutT, 512, 512);
  k_convert_x<<<(nx / 4 + 255) / 256, 256, 0, stream>>>(x, xb, nx);
  k_qkv_gemm<<<dim3(4, 6, 32), 256, 0, stream>>>(xb, WqkvT, q, kmat, vT);
  k_attn<<<dim3(32, 8, 32), 32, 0, stream>>>(q, kmat, vT, ao);
  k_out_gemm<<<dim3(4, 2, 32), 256, 0, stream>>>(ao, WoutT, b_out, out);
}
